// NetVLADLoupe_45354854645906
// MI455X (gfx1250) — matrix operand order repaired and checked by reading, not compiled, not run on hardware
//
#include <hip/hip_runtime.h>

#define DEVINL __device__ __forceinline__

typedef __attribute__((ext_vector_type(4)))  float           f4_t;
typedef __attribute__((ext_vector_type(8)))  float           v8f;
typedef __attribute__((ext_vector_type(16))) __bf16          v16bf;
typedef __attribute__((ext_vector_type(8)))  unsigned short  us8;
typedef __attribute__((ext_vector_type(4)))  unsigned short  us4;

union Frag16 {
  v16bf          v;
  f4_t           f[2];
  unsigned short u[16];
};

static constexpr int NP    = 64000;   // points
static constexpr int FEATD = 256;     // features
static constexpr int KC    = 64;      // clusters
static constexpr int BSEG  = 16;      // segments
static constexpr int OUTD  = 1024;    // output dim
static constexpr int PS    = 64032;   // padded point-stride (halfs) for featT/actT
static constexpr int KSPLIT = 16;     // K-splits per segment in GEMM2 (256 WGs total)

// ---- workspace layout (bytes) ----
static constexpr size_t FEATT_OFF  = 0;                                    // bf16 [256][PS]
static constexpr size_t ACTT_OFF   = FEATT_OFF + (size_t)FEATD * PS * 2;   // bf16 [64][PS]
static constexpr size_t ACTRAW_OFF = ACTT_OFF  + (size_t)KC * PS * 2;      // f32  [NP][64]
static constexpr size_t BSWZ_OFF   = ACTRAW_OFF + (size_t)NP * KC * 4;     // bf16 swizzled cw (32KB)
static constexpr size_t VPART_OFF  = BSWZ_OFF + 32768;                     // f32 [KSPLIT][16][256][64]
static constexpr size_t VLADN_OFF  = VPART_OFF + (size_t)KSPLIT*BSEG*FEATD*KC*4; // f32 [16][16384]
static constexpr size_t OUTRAW_OFF = VLADN_OFF + (size_t)BSEG*FEATD*KC*4;  // f32 [16][1024]
static constexpr size_t PARTS_OFF  = OUTRAW_OFF + (size_t)BSEG*OUTD*4;     // f32 [64][64]
static constexpr size_t PARTQ_OFF  = PARTS_OFF + 64*64*4;                  // f32 [64][64]
static constexpr size_t BNSC_OFF   = PARTQ_OFF + 64*64*4;                  // f32 [64]
static constexpr size_t BNSH_OFF   = BNSC_OFF + 256;                       // f32 [64]
static constexpr size_t ASUM_OFF   = BNSH_OFF + 256;                       // f32 [16][64]
static constexpr size_t SEGO_OFF   = ASUM_OFF + 4096;                      // int [17]

DEVINL unsigned short f2bf(float f) {            // round-to-nearest-even f32->bf16
  unsigned int x = __builtin_bit_cast(unsigned int, f);
  unsigned int r = x + 0x7FFFu + ((x >> 16) & 1u);
  return (unsigned short)(r >> 16);
}
DEVINL float bf2f(unsigned short u) {
  unsigned int x = ((unsigned int)u) << 16;
  return __builtin_bit_cast(float, x);
}
DEVINL v8f v8f_zero() {
  v8f z;
#pragma unroll
  for (int i = 0; i < 8; ++i) z[i] = 0.f;
  return z;
}
DEVINL v8f wmma_bf16(v16bf a, v16bf b, v8f c) {
  return __builtin_amdgcn_wmma_f32_16x16x32_bf16(false, a, false, b, (short)0, c, false, false);
}

// ---------------------------------------------------------------------------
// K0: segment offsets, swizzled bf16 cluster weights, zero transposed pads
// ---------------------------------------------------------------------------
__global__ __launch_bounds__(256) void k0_prep(const int* __restrict__ coords,
                                               const float* __restrict__ cw,
                                               char* __restrict__ ws) {
  int gid = blockIdx.x * 256 + threadIdx.x;
  int* sego = (int*)(ws + SEGO_OFF);
  unsigned short* bswz  = (unsigned short*)(ws + BSWZ_OFF);
  unsigned short* featT = (unsigned short*)(ws + FEATT_OFF);
  unsigned short* actT  = (unsigned short*)(ws + ACTT_OFF);

  if (gid < NP) {
    int s  = coords[4 * gid];
    int sp = (gid > 0) ? coords[4 * (gid - 1)] : -1;
    for (int b = sp + 1; b <= s; ++b) sego[b] = gid;
    if (gid == NP - 1)
      for (int b = s + 1; b <= BSEG; ++b) sego[b] = NP;
  }
  // B-fragment swizzle for GEMM1: [ks(8)][nt(4)][lane(32)][e(16)]
  if (gid < 8 * 4 * 32 * 16) {
    int e  = gid & 15;
    int l  = (gid >> 4) & 31;
    int nt = (gid >> 9) & 3;
    int ks = gid >> 11;
    int hi = l >> 4;
    int f  = ks * 32 + hi * 16 + e;       // K (feature) index
    int n  = nt * 16 + (l & 15);          // cluster column
    bswz[gid] = f2bf(cw[f * KC + n]);
  }
  if (gid < FEATD * 32) { int f = gid >> 5, o = gid & 31; featT[(size_t)f * PS + NP + o] = 0; }
  if (gid < KC    * 32) { int k = gid >> 5, o = gid & 31; actT [(size_t)k * PS + NP + o] = 0; }
}

// ---------------------------------------------------------------------------
// K1: act_raw = feat @ cw  (bf16 WMMA)  + write transposed bf16 featT
//     grid: 1000 x 128 (4 waves), 64 rows / WG
// ---------------------------------------------------------------------------
__global__ __launch_bounds__(128) void k1_gemm1(const float* __restrict__ feat,
                                                char* __restrict__ ws) {
  __shared__ unsigned short tile[64 * 264];     // 64 rows x 256 cols bf16, padded
  const int tid = threadIdx.x;
  const int r0  = blockIdx.x * 64;
  unsigned short* featT = (unsigned short*)(ws + FEATT_OFF);
  const unsigned short* bswz = (const unsigned short*)(ws + BSWZ_OFF);
  float* actraw = (float*)(ws + ACTRAW_OFF);

  // load 64x256 f32 -> bf16 LDS tile (coalesced float4)
  const f4_t* f4p = (const f4_t*)feat;
#pragma unroll
  for (int j = 0; j < 32; ++j) {
    int idx4 = tid + j * 128;                    // 4096 float4 total
    int row  = idx4 >> 6;
    int c4   = idx4 & 63;
    f4_t d = f4p[(size_t)(r0 + row) * 64 + c4];
    us4 p;
#pragma unroll
    for (int i = 0; i < 4; ++i) p[i] = f2bf(d[i]);
    *(us4*)&tile[row * 264 + c4 * 4] = p;
  }
  __syncthreads();

  // transposed coalesced write of featT[f][r0..r0+63]
#pragma unroll
  for (int j = 0; j < 16; ++j) {
    int idx8 = tid + j * 128;                    // 2048 8-half units
    int f    = idx8 >> 3;
    int nch  = idx8 & 7;
    us8 p;
#pragma unroll
    for (int i = 0; i < 8; ++i) p[i] = tile[(nch * 8 + i) * 264 + f];
    *(us8*)(featT + (size_t)f * PS + r0 + nch * 8) = p;
  }

  // WMMA: wave w handles rows r0+16w .. +16, all 64 columns
  const int wave = tid >> 5, lane = tid & 31, hi = lane >> 4, lo = lane & 15;
  const int lm = wave * 16 + lo;
  v8f acc[4];
#pragma unroll
  for (int nt = 0; nt < 4; ++nt) acc[nt] = v8f_zero();

#pragma unroll
  for (int ks = 0; ks < 8; ++ks) {
    Frag16 a;
    a.f[0] = *(const f4_t*)&tile[lm * 264 + ks * 32 + hi * 16];
    a.f[1] = *(const f4_t*)&tile[lm * 264 + ks * 32 + hi * 16 + 8];
#pragma unroll
    for (int nt = 0; nt < 4; ++nt) {
      Frag16 b;
      const unsigned short* bp = bswz + ((size_t)((ks * 4 + nt) * 32 + lane)) * 16;
      b.f[0] = *(const f4_t*)bp;
      b.f[1] = *(const f4_t*)(bp + 8);
      acc[nt] = wmma_bf16(a.v, b.v, acc[nt]);
    }
  }
#pragma unroll
  for (int nt = 0; nt < 4; ++nt) {
#pragma unroll
    for (int r = 0; r < 8; ++r) {
      actraw[(size_t)(r0 + wave * 16 + hi * 8 + r) * KC + nt * 16 + lo] = acc[nt][r];
    }
  }
}

// ---------------------------------------------------------------------------
// K2a/K2b: deterministic BN1 stats (mean/var over N per column)
// ---------------------------------------------------------------------------
__global__ __launch_bounds__(256) void k2a_stats(char* __restrict__ ws) {
  __shared__ float red[256][8];
  const float* actraw = (const float*)(ws + ACTRAW_OFF);
  float* parts = (float*)(ws + PARTS_OFF);
  float* partq = (float*)(ws + PARTQ_OFF);
  int t = threadIdx.x, r0 = blockIdx.x * 1000;
  int cg = t & 15, rsub = t >> 4;
  float s[4] = {0, 0, 0, 0}, q[4] = {0, 0, 0, 0};
  for (int pass = 0; pass < 63; ++pass) {
    int row = rsub + pass * 16;
    if (row >= 1000) break;
    f4_t d = *(const f4_t*)(actraw + (size_t)(r0 + row) * 64 + cg * 4);
#pragma unroll
    for (int i = 0; i < 4; ++i) { s[i] += d[i]; q[i] += d[i] * d[i]; }
  }
#pragma unroll
  for (int i = 0; i < 4; ++i) { red[t][i] = s[i]; red[t][4 + i] = q[i]; }
  __syncthreads();
  if (t < 64) {
    int grp = t >> 2, slot = t & 3;
    float S = 0.f, Q = 0.f;
    for (int i = 0; i < 16; ++i) { S += red[grp + 16 * i][slot]; Q += red[grp + 16 * i][4 + slot]; }
    parts[blockIdx.x * 64 + t] = S;
    partq[blockIdx.x * 64 + t] = Q;
  }
}
__global__ void k2b_finalize(const float* __restrict__ g, const float* __restrict__ bta,
                             char* __restrict__ ws) {
  int c = threadIdx.x;
  if (c >= KC) return;
  const float* parts = (const float*)(ws + PARTS_OFF);
  const float* partq = (const float*)(ws + PARTQ_OFF);
  float S = 0.f, Q = 0.f;
  for (int i = 0; i < 64; ++i) { S += parts[i * 64 + c]; Q += partq[i * 64 + c]; }
  float mean = S / (float)NP;
  float var  = Q / (float)NP - mean * mean;
  float rstd = rsqrtf(var + 1e-5f);
  float sc = rstd * g[c];
  ((float*)(ws + BNSC_OFF))[c] = sc;
  ((float*)(ws + BNSH_OFF))[c] = bta[c] - mean * sc;
}

// ---------------------------------------------------------------------------
// K3: BN + softmax -> transposed bf16 actT.  grid 1000 x 256, 64 rows / WG
// ---------------------------------------------------------------------------
__global__ __launch_bounds__(256) void k3_softmax(char* __restrict__ ws) {
  __shared__ float lt[64 * 65];
  __shared__ unsigned short ldsT[64 * 72];
  const float* actraw = (const float*)(ws + ACTRAW_OFF);
  const float* bnsc = (const float*)(ws + BNSC_OFF);
  const float* bnsh = (const float*)(ws + BNSH_OFF);
  unsigned short* actT = (unsigned short*)(ws + ACTT_OFF);
  int tid = threadIdx.x, r0 = blockIdx.x * 64;
#pragma unroll
  for (int j = 0; j < 16; ++j) {
    int idx = tid + j * 256;
    int row = idx >> 6, c = idx & 63;
    float x = actraw[(size_t)(r0 + row) * 64 + c];
    lt[row * 65 + c] = x * bnsc[c] + bnsh[c];
  }
  __syncthreads();
  int row = tid >> 2, sub = tid & 3;
  const float* rp = &lt[row * 65 + sub * 16];
  float m = -1e30f;
#pragma unroll
  for (int i = 0; i < 16; ++i) m = fmaxf(m, rp[i]);
  m = fmaxf(m, __shfl_xor(m, 1, 32));
  m = fmaxf(m, __shfl_xor(m, 2, 32));
  float ev[16], s = 0.f;
#pragma unroll
  for (int i = 0; i < 16; ++i) { ev[i] = __expf(rp[i] - m); s += ev[i]; }
  s += __shfl_xor(s, 1, 32);
  s += __shfl_xor(s, 2, 32);
  float inv = 1.0f / s;
#pragma unroll
  for (int i = 0; i < 16; ++i) ldsT[(sub * 16 + i) * 72 + row] = f2bf(ev[i] * inv);
  __syncthreads();
#pragma unroll
  for (int j = 0; j < 2; ++j) {
    int idx8 = tid + j * 256;
    int k = idx8 >> 3, nch = idx8 & 7;
    us8 d = *(const us8*)&ldsT[k * 72 + nch * 8];
    *(us8*)(actT + (size_t)k * PS + r0 + nch * 8) = d;
  }
}

// ---------------------------------------------------------------------------
// K4: a_sum[b][k] = sum over segment b of act[n][k].  1024 waves.
// ---------------------------------------------------------------------------
__global__ __launch_bounds__(256) void k4_asum(char* __restrict__ ws) {
  int W = blockIdx.x * 8 + (threadIdx.x >> 5);
  int lane = threadIdx.x & 31;
  int b = W >> 6, k = W & 63;
  const int* sego = (const int*)(ws + SEGO_OFF);
  const unsigned short* actT = (const unsigned short*)(ws + ACTT_OFF);
  int st = sego[b], en = sego[b + 1];
  float s = 0.f;
  for (int n = st + lane; n < en; n += 32) s += bf2f(actT[(size_t)k * PS + n]);
#pragma unroll
  for (int msk = 16; msk >= 1; msk >>= 1) s += __shfl_xor(s, msk, 32);
  if (lane == 0) ((float*)(ws + ASUM_OFF))[b * 64 + k] = s;
}

// ---------------------------------------------------------------------------
// K5: per-segment GEMM  vlad_part[s][b] += feat_b^T @ act_b  (bf16 WMMA)
//     grid 256 = 16 segs x 16 K-splits; 8 waves; each wave 32x64 output
// ---------------------------------------------------------------------------
__global__ __launch_bounds__(256) void k5_gemm2(char* __restrict__ ws) {
  int sp = blockIdx.x & (KSPLIT - 1), b = blockIdx.x / KSPLIT;
  int w = threadIdx.x >> 5, lane = threadIdx.x & 31, hi = lane >> 4, lo = lane & 15;
  const int* sego = (const int*)(ws + SEGO_OFF);
  const unsigned short* featT = (const unsigned short*)(ws + FEATT_OFF);
  const unsigned short* actT  = (const unsigned short*)(ws + ACTT_OFF);
  int st = sego[b], en = sego[b + 1];
  int base = st & ~31;
  int TB = (en - base + 31) >> 5;
  int bps = (TB + KSPLIT - 1) / KSPLIT;
  int b0 = sp * bps;
  int b1 = (b0 + bps < TB) ? (b0 + bps) : TB;

  v8f acc[2][4];
#pragma unroll
  for (int mt = 0; mt < 2; ++mt)
#pragma unroll
    for (int nt = 0; nt < 4; ++nt) acc[mt][nt] = v8f_zero();

  for (int blk = b0; blk < b1; ++blk) {
    int kb = base + blk * 32;
    if (blk + 1 < b1)
      __builtin_prefetch((const void*)(featT + (size_t)(w * 32 + lo) * PS + kb + 32), 0, 0);
    Frag16 a[2];
#pragma unroll
    for (int mt = 0; mt < 2; ++mt) {
      int f = w * 32 + mt * 16 + lo;
      const unsigned short* p = featT + (size_t)f * PS + kb + hi * 16;
      a[mt].f[0] = *(const f4_t*)p;
      a[mt].f[1] = *(const f4_t*)(p + 8);
    }
    Frag16 bf[4];
    bool full = (kb >= st) && (kb + 32 <= en);
    if (full) {
#pragma unroll
      for (int nt = 0; nt < 4; ++nt) {
        const unsigned short* p = actT + (size_t)(nt * 16 + lo) * PS + kb + hi * 16;
        bf[nt].f[0] = *(const f4_t*)p;
        bf[nt].f[1] = *(const f4_t*)(p + 8);
      }
    } else {
#pragma unroll
      for (int nt = 0; nt < 4; ++nt) {
#pragma unroll
        for (int e = 0; e < 16; ++e) {
          int n = kb + hi * 16 + e;
          bf[nt].u[e] = (n >= st && n < en)
                            ? actT[(size_t)(nt * 16 + lo) * PS + n]
                            : (unsigned short)0;
        }
      }
    }
#pragma unroll
    for (int mt = 0; mt < 2; ++mt)
#pragma unroll
      for (int nt = 0; nt < 4; ++nt) acc[mt][nt] = wmma_bf16(a[mt].v, bf[nt].v, acc[mt][nt]);
  }

  float* vp = (float*)(ws + VPART_OFF) + (size_t)(sp * BSEG + b) * FEATD * KC;
#pragma unroll
  for (int mt = 0; mt < 2; ++mt)
#pragma unroll
    for (int nt = 0; nt < 4; ++nt)
#pragma unroll
      for (int r = 0; r < 8; ++r)
        vp[(size_t)(w * 32 + mt * 16 + hi * 8 + r) * KC + nt * 16 + lo] = acc[mt][nt][r];
}

// ---------------------------------------------------------------------------
// K6: vlad = sum(parts) - a_sum*cw2; l2norm(axis=f); l2norm(flat row) -> vlad_n
//     grid 16 (one per b) x 256 (thread == f)
// ---------------------------------------------------------------------------
__global__ __launch_bounds__(256) void k6_vlad(const float* __restrict__ cw2,
                                               char* __restrict__ ws) {
  __shared__ float wredc[8][64];
  __shared__ float wredr[8];
  __shared__ float invc[64];
  __shared__ float sb_sh;
  int b = blockIdx.x, t = threadIdx.x;
  int wave = t >> 5, lane = t & 31;
  const float* vp = (const float*)(ws + VPART_OFF);
  const float* asum = (const float*)(ws + ASUM_OFF) + b * 64;

  float v[64];
#pragma unroll
  for (int k = 0; k < 64; ++k) v[k] = 0.f;
  for (int s = 0; s < KSPLIT; ++s) {
    const f4_t* p = (const f4_t*)(vp + ((size_t)(s * BSEG + b) * FEATD + t) * KC);
#pragma unroll
    for (int q = 0; q < 16; ++q) {
      f4_t d = p[q];
#pragma unroll
      for (int i = 0; i < 4; ++i) v[4 * q + i] += d[i];
    }
  }
#pragma unroll
  for (int k = 0; k < 64; ++k) v[k] -= asum[k] * cw2[(size_t)t * 64 + k];

  // deterministic column (over f) sum of squares: butterfly within wave, then LDS
  float v2[64];
#pragma unroll
  for (int k = 0; k < 64; ++k) v2[k] = v[k] * v[k];
#pragma unroll
  for (int msk = 1; msk < 32; msk <<= 1)
#pragma unroll
    for (int k = 0; k < 64; ++k) v2[k] += __shfl_xor(v2[k], msk, 32);
  if (lane == 0)
#pragma unroll
    for (int k = 0; k < 64; ++k) wredc[wave][k] = v2[k];
  __syncthreads();
  if (t < 64) {
    float cs = 0.f;
#pragma unroll
    for (int wv = 0; wv < 8; ++wv) cs += wredc[wv][t];
    invc[t] = 1.0f / fmaxf(sqrtf(cs), 1e-12f);
  }
  __syncthreads();

  float rs = 0.f;
#pragma unroll
  for (int k = 0; k < 64; ++k) {
    float x = v[k] * invc[k];
    v[k] = x;
    rs += x * x;
  }
#pragma unroll
  for (int msk = 1; msk < 32; msk <<= 1) rs += __shfl_xor(rs, msk, 32);
  if (lane == 0) wredr[wave] = rs;
  __syncthreads();
  if (t == 0) {
    float tot = 0.f;
#pragma unroll
    for (int wv = 0; wv < 8; ++wv) tot += wredr[wv];
    sb_sh = 1.0f / fmaxf(sqrtf(tot), 1e-12f);
  }
  __syncthreads();
  float sb = sb_sh;
  float* vn = (float*)(ws + VLADN_OFF) + (size_t)b * (FEATD * KC) + (size_t)t * 64;
#pragma unroll
  for (int k = 0; k < 64; ++k) vn[k] = v[k] * sb;
}

// ---------------------------------------------------------------------------
// K7: out_raw = vlad_n @ W1  (bf16 WMMA, convert W1 f32->bf16 in regs)
//     grid 64 (N-tiles of 16) x 1024 (32 waves); waves split K=16384 by 512
// ---------------------------------------------------------------------------
__global__ __launch_bounds__(1024) void k7_gemm3(const float* __restrict__ hw,
                                                 char* __restrict__ ws) {
  __shared__ float red[32 * 256];
  int nt = blockIdx.x;
  int w = threadIdx.x >> 5, lane = threadIdx.x & 31, hi = lane >> 4, lo = lane & 15;
  const float* vn = (const float*)(ws + VLADN_OFF);
  int n = nt * 16 + lo;
  v8f acc = v8f_zero();

  for (int step = 0; step < 16; ++step) {
    int kb = w * 512 + step * 32;
    if (step + 1 < 16)
      __builtin_prefetch((const void*)(hw + (size_t)(kb + 32 + hi * 16) * OUTD + n), 0, 0);
    Frag16 a;
    const float* ap = vn + (size_t)lo * (FEATD * KC) + kb + hi * 16;
    f4_t r0 = *(const f4_t*)ap;
    f4_t r1 = *(const f4_t*)(ap + 4);
    f4_t r2 = *(const f4_t*)(ap + 8);
    f4_t r3 = *(const f4_t*)(ap + 12);
#pragma unroll
    for (int i = 0; i < 4; ++i) {
      a.u[i]      = f2bf(r0[i]);
      a.u[4 + i]  = f2bf(r1[i]);
      a.u[8 + i]  = f2bf(r2[i]);
      a.u[12 + i] = f2bf(r3[i]);
    }
    Frag16 bf;
#pragma unroll
    for (int e = 0; e < 16; ++e)
      bf.u[e] = f2bf(hw[(size_t)(kb + hi * 16 + e) * OUTD + n]);
    acc = wmma_bf16(a.v, bf.v, acc);
  }
#pragma unroll
  for (int r = 0; r < 8; ++r) red[w * 256 + (hi * 8 + r) * 16 + lo] = acc[r];
  __syncthreads();
  int t = threadIdx.x;
  if (t < 256) {
    float sm = 0.f;
#pragma unroll
    for (int wv = 0; wv < 32; ++wv) sm += red[wv * 256 + t];
    ((float*)(ws + OUTRAW_OFF))[(size_t)(t >> 4) * OUTD + nt * 16 + (t & 15)] = sm;
  }
}

// ---------------------------------------------------------------------------
// K8: BN2 over B=16 rows per column -> d_out
// ---------------------------------------------------------------------------
__global__ __launch_bounds__(256) void k8_bn2(const float* __restrict__ g,
                                              const float* __restrict__ bta,
                                              char* __restrict__ ws,
                                              float* __restrict__ out) {
  int c = blockIdx.x * 256 + threadIdx.x;
  if (c >= OUTD) return;
  const float* oraw = (const float*)(ws + OUTRAW_OFF);
  float x[16], s = 0.f;
#pragma unroll
  for (int r = 0; r < 16; ++r) { x[r] = oraw[(size_t)r * OUTD + c]; s += x[r]; }
  float mean = s * (1.0f / 16.0f);
  float q = 0.f;
#pragma unroll
  for (int r = 0; r < 16; ++r) { float d = x[r] - mean; q += d * d; }
  float rstd = rsqrtf(q * (1.0f / 16.0f) + 1e-5f);
  float sc = rstd * g[c], sh = bta[c] - mean * sc;
#pragma unroll
  for (int r = 0; r < 16; ++r) out[(size_t)r * OUTD + c] = x[r] * sc + sh;
}

// ---------------------------------------------------------------------------
extern "C" void kernel_launch(void* const* d_in, const int* in_sizes, int n_in,
                              void* d_out, int out_size, void* d_ws, size_t ws_size,
                              hipStream_t stream) {
  const float* feat = (const float*)d_in[0];
  const int*   crd  = (const int*)d_in[1];
  const float* cw   = (const float*)d_in[2];
  const float* cw2  = (const float*)d_in[3];
  const float* hw   = (const float*)d_in[4];
  const float* g1   = (const float*)d_in[5];
  const float* b1   = (const float*)d_in[6];
  const float* g2   = (const float*)d_in[7];
  const float* b2   = (const float*)d_in[8];
  char* ws = (char*)d_ws;
  float* out = (float*)d_out;
  (void)in_sizes; (void)n_in; (void)out_size; (void)ws_size;

  k0_prep     <<<250,  256,  0, stream>>>(crd, cw, ws);
  k1_gemm1    <<<1000, 128,  0, stream>>>(feat, ws);
  k2a_stats   <<<64,   256,  0, stream>>>(ws);
  k2b_finalize<<<1,     64,  0, stream>>>(g1, b1, ws);
  k3_softmax  <<<1000, 256,  0, stream>>>(ws);
  k4_asum     <<<128,  256,  0, stream>>>(ws);
  k5_gemm2    <<<BSEG * KSPLIT, 256, 0, stream>>>(ws);
  k6_vlad     <<<16,   256,  0, stream>>>(cw2, ws);
  k7_gemm3    <<<64,   1024, 0, stream>>>(hw, ws);
  k8_bn2      <<<4,    256,  0, stream>>>(g2, b2, ws, out);
}
